// RN_64098091926199
// MI455X (gfx1250) — compile-verified, matrix operand order checked by reading, not run on hardware
//
#include <hip/hip_runtime.h>
#include <hip/hip_bf16.h>

typedef __bf16 v16bf __attribute__((ext_vector_type(16)));
typedef float  v8f   __attribute__((ext_vector_type(8)));

// ---------- helpers ----------
__device__ __forceinline__ unsigned short f32_to_bf16(float f) {
    unsigned int u = __float_as_uint(f);
    unsigned int r = u + 0x7FFFu + ((u >> 16) & 1u);   // round-to-nearest-even
    return (unsigned short)(r >> 16);
}

// forward K mapping for 16-bit A/B fragments (ISA 7.12.2, 16-bit A 16x32 table)
// lane-half + element e (0..15) -> K (0..31)
__device__ __forceinline__ int kmap_fwd(int e, int half) {
    int p = e >> 1, o = e & 1;
    return 2 * p + o + 8 * ((p >> 2) + half);
}
// inverse: kr (0..31) -> (half, e)
__device__ __forceinline__ void kmap_inv(int kr, int& half, int& e) {
    int o = kr & 1, p;
    if (kr < 8)       { half = 0; p = kr >> 1; }
    else if (kr < 16) { half = 1; p = (kr - 8) >> 1; }
    else if (kr < 24) { half = 0; p = ((kr - 16) >> 1) + 4; }
    else              { half = 1; p = ((kr - 24) >> 1) + 4; }
    e = (p << 1) | o;
}
// offset (in ushorts) of element (m,k) in the LDS A-fragment buffer
// layout: [kk(16)][mt(2)][lane(32)][e(16)]
__device__ __forceinline__ int afrag_off(int m, int k) {
    int kk = k >> 5, kr = k & 31, half, e;
    kmap_inv(kr, half, e);
    return (((kk * 2 + (m >> 4)) * 32) + (half * 16 + (m & 15))) * 16 + e;
}

// ---------- conv 3x3 stride 2 pad 1 + ReLU ----------
__global__ void conv3x3s2_relu(const float* __restrict__ x, const float* __restrict__ w,
                               const float* __restrict__ bias, float* __restrict__ y,
                               int N, int Cin, int Hin, int Win, int Cout, int Hout, int Wout) {
    int idx = blockIdx.x * blockDim.x + threadIdx.x;
    int total = N * Cout * Hout * Wout;
    if (idx >= total) return;
    int wo = idx % Wout; int t = idx / Wout;
    int ho = t % Hout;   t /= Hout;
    int co = t % Cout;   int n = t / Cout;
    float s = bias[co];
    const float* wb = w + (size_t)co * Cin * 9;
    const float* xb = x + (size_t)n * Cin * Hin * Win;
    for (int ci = 0; ci < Cin; ++ci) {
        const float* xc = xb + (size_t)ci * Hin * Win;
        const float* wc = wb + ci * 9;
#pragma unroll
        for (int kh = 0; kh < 3; ++kh) {
            int hi = ho * 2 - 1 + kh;
            if (hi < 0 || hi >= Hin) continue;
#pragma unroll
            for (int kw = 0; kw < 3; ++kw) {
                int wi = wo * 2 - 1 + kw;
                if (wi < 0 || wi >= Win) continue;
                s += xc[hi * Win + wi] * wc[kh * 3 + kw];
            }
        }
    }
    y[idx] = s > 0.f ? s : 0.f;
}

// ---------- batchnorm: per-channel mean / invstd ----------
__global__ void bn_stats(const float* __restrict__ x, float* __restrict__ stats,
                         int N, int C, int HW) {
    int ch = blockIdx.x, tid = threadIdx.x;
    int total = N * HW;
    float s = 0.f, sq = 0.f;
    for (int i = tid; i < total; i += 256) {
        int n = i / HW, r = i - n * HW;
        float v = x[((size_t)n * C + ch) * HW + r];
        s += v; sq += v * v;
    }
    __shared__ float sh[256], sh2[256];
    sh[tid] = s; sh2[tid] = sq; __syncthreads();
    for (int st = 128; st > 0; st >>= 1) {
        if (tid < st) { sh[tid] += sh[tid + st]; sh2[tid] += sh2[tid + st]; }
        __syncthreads();
    }
    if (tid == 0) {
        float mean = sh[0] / total;
        float var  = sh2[0] / total - mean * mean;
        stats[ch]     = mean;
        stats[C + ch] = rsqrtf(var + 1e-5f);
    }
}

__global__ void bn_apply(float* __restrict__ x, const float* __restrict__ stats,
                         const float* __restrict__ g, const float* __restrict__ bb,
                         int C, int HW, int total) {
    int idx = blockIdx.x * blockDim.x + threadIdx.x;
    if (idx >= total) return;
    int c = (idx / HW) % C;
    x[idx] = (x[idx] - stats[c]) * stats[C + c] * g[c] + bb[c];
}

// ---------- ti / tj projections (258 -> 512), coords fused ----------
__global__ void titj_kernel(const float* __restrict__ x4, const float* __restrict__ g1w,
                            float* __restrict__ ti, float* __restrict__ tj) {
    int idx = blockIdx.x * blockDim.x + threadIdx.x;     // 128*25*512
    if (idx >= 128 * 25 * 512) return;
    int d = idx & 511;
    int t = idx >> 9;
    int p = t % 25, b = t / 25;
    float si = 0.f, sj = 0.f;
    const float* xb = x4 + (size_t)b * 256 * 25 + p;     // channel stride 25
    for (int c = 0; c < 256; ++c) {
        float xv = xb[c * 25];
        sj += xv * g1w[c * 512 + d];
        si += xv * g1w[(258 + c) * 512 + d];
    }
    float c0 = (float)p * 0.1f - 1.0f;                   // (p/5.0 - 2)/2
    float c1 = (float)(p % 5) * 0.5f - 1.0f;             // (p%5 - 2)/2
    sj += c0 * g1w[256 * 512 + d] + c1 * g1w[257 * 512 + d];
    si += c0 * g1w[(258 + 256) * 512 + d] + c1 * g1w[(258 + 257) * 512 + d];
    tj[idx] = sj; ti[idx] = si;
}

__global__ void tq_kernel(const float* __restrict__ qst, const float* __restrict__ g1w,
                          const float* __restrict__ g1b, float* __restrict__ tq) {
    int idx = blockIdx.x * blockDim.x + threadIdx.x;     // 128*512
    if (idx >= 128 * 512) return;
    int d = idx & 511, b = idx >> 9;
    float s = g1b[d];
    for (int c = 0; c < 11; ++c) s += qst[b * 11 + c] * g1w[(516 + c) * 512 + d];
    tq[idx] = s;
}

// ---------- pack g2/g3/g4 weights into per-lane-contiguous bf16 B-fragments ----------
__global__ void pack_w(const float* __restrict__ w2, const float* __restrict__ w3,
                       const float* __restrict__ w4, unsigned short* __restrict__ Wp) {
    int idx = blockIdx.x * blockDim.x + threadIdx.x;     // 3 * 2^18
    if (idx >= 3 * 512 * 512) return;
    int wsel = idx >> 18;
    int rem  = idx & ((1 << 18) - 1);
    int e    = rem & 15;
    int lane = (rem >> 4) & 31;
    int fragidx = rem >> 9;
    int nt = fragidx & 31, kk = fragidx >> 5;
    int half = lane >> 4;
    int n = nt * 16 + (lane & 15);
    int k = kk * 32 + kmap_fwd(e, half);
    const float* w = wsel == 0 ? w2 : (wsel == 1 ? w3 : w4);
    Wp[idx] = f32_to_bf16(w[k * 512 + n]);
}

__global__ void zero_f(float* __restrict__ p, int n) {
    int idx = blockIdx.x * blockDim.x + threadIdx.x;
    if (idx < n) p[idx] = 0.f;
}

// ---------- fused relation-MLP: h0 build -> 3x (512x512 WMMA GEMM + ReLU) -> sum ----------
__global__ __launch_bounds__(256)
void rn_fused(const float* __restrict__ ti, const float* __restrict__ tj,
              const float* __restrict__ tq, const unsigned short* __restrict__ Wp,
              const float* __restrict__ b2, const float* __restrict__ b3,
              const float* __restrict__ b4, float* __restrict__ xg) {
    __shared__ unsigned short frag[16 * 2 * 32 * 16];    // 32 KB A-fragment buffer (32 x 512 bf16)
    __shared__ float accs[512];                           // column accumulator for x_g

    const int b    = blockIdx.x;      // 0..127
    const int tile = blockIdx.y;      // 0..19 (20 * 32 >= 625 pair rows)
    const int tid  = threadIdx.x;
    const int lane = tid & 31;
    const int wave = tid >> 5;
    const int row0 = tile * 32;

    for (int i = tid; i < 512; i += 256) accs[i] = 0.f;

    // stage 0: h0 = relu(ti[b,i] + tj[b,j] + tq[b]), stored directly in A-frag layout
    const float* tib = ti + (size_t)b * 25 * 512;
    const float* tjb = tj + (size_t)b * 25 * 512;
    const float* tqb = tq + (size_t)b * 512;
    for (int idx = tid; idx < 32 * 512; idx += 256) {
        int m = idx >> 9, k = idx & 511;
        int r = row0 + m; if (r > 624) r = 624;           // pad rows: garbage, masked at final sum
        int i_ = r / 25, j_ = r - i_ * 25;
        float v = tib[i_ * 512 + k] + tjb[j_ * 512 + k] + tqb[k];
        v = v > 0.f ? v : 0.f;
        frag[afrag_off(m, k)] = f32_to_bf16(v);
    }
    __syncthreads();

    for (int layer = 0; layer < 3; ++layer) {
        const unsigned short* W = Wp + (size_t)layer * 512 * 512;
        const float* bias = layer == 0 ? b2 : (layer == 1 ? b3 : b4);

        v8f acc[8];
        // each wave owns 8 of the 64 output 16x16 tiles (mt in 0..1, nt in 0..31)
        for (int t = 0; t < 8; ++t) {
            int idx = wave * 8 + t;
            int mt = idx >> 5, nt = idx & 31;
            v8f c = {0.f, 0.f, 0.f, 0.f, 0.f, 0.f, 0.f, 0.f};
#pragma unroll
            for (int kk = 0; kk < 16; ++kk) {
                v16bf a = *reinterpret_cast<const v16bf*>(&frag[((kk * 2 + mt) * 32 + lane) * 16]);
                v16bf bm = *reinterpret_cast<const v16bf*>(&W[(((kk << 5) + nt) * 32 + lane) * 16]);
                c = __builtin_amdgcn_wmma_f32_16x16x32_bf16(false, a, false, bm,
                                                            (short)0, c, false, false);
            }
            acc[t] = c;
        }
        __syncthreads();   // all WMMA reads of frag complete -> safe to overwrite in place

        if (layer < 2) {
            for (int t = 0; t < 8; ++t) {
                int idx = wave * 8 + t;
                int mt = idx >> 5, nt = idx & 31;
                int n = nt * 16 + (lane & 15);            // fixed per lane (C/D layout)
                float bia = bias[n];
                int kk2 = n >> 5, half_k, e;
                kmap_inv(n & 31, half_k, e);
#pragma unroll
                for (int v = 0; v < 8; ++v) {
                    int m = mt * 16 + v + (lane >> 4) * 8;
                    float val = acc[t][v] + bia;
                    val = val > 0.f ? val : 0.f;
                    frag[((kk2 * 2 + (m >> 4)) * 32 + (half_k * 16 + (m & 15))) * 16 + e]
                        = f32_to_bf16(val);
                }
            }
            __syncthreads();
        } else {
            for (int t = 0; t < 8; ++t) {
                int idx = wave * 8 + t;
                int mt = idx >> 5, nt = idx & 31;
                int n = nt * 16 + (lane & 15);
                float bia = bias[n];
                float s = 0.f;
#pragma unroll
                for (int v = 0; v < 8; ++v) {
                    int m = mt * 16 + v + (lane >> 4) * 8;
                    if (row0 + m < 625) {
                        float val = acc[t][v] + bia;
                        s += val > 0.f ? val : 0.f;
                    }
                }
                atomicAdd(&accs[n], s);                   // LDS f32 atomic
            }
            __syncthreads();
            for (int i = tid; i < 512; i += 256)
                unsafeAtomicAdd(&xg[(size_t)b * 512 + i], accs[i]);   // global_atomic_add_f32
        }
    }
}

// ---------- fc head ----------
__global__ void fc_kernel(const float* __restrict__ in, const float* __restrict__ W,
                          const float* __restrict__ bias, float* __restrict__ out,
                          int B, int K, int Nn, int do_relu) {
    int idx = blockIdx.x * blockDim.x + threadIdx.x;
    if (idx >= B * Nn) return;
    int n = idx % Nn, b = idx / Nn;
    float s = bias[n];
    for (int k = 0; k < K; ++k) s += in[(size_t)b * K + k] * W[(size_t)k * Nn + n];
    if (do_relu) s = s > 0.f ? s : 0.f;
    out[idx] = s;
}

__global__ void log_softmax10(const float* __restrict__ y, float* __restrict__ out) {
    int b = blockIdx.x * blockDim.x + threadIdx.x;
    if (b >= 128) return;
    float mx = -3.4e38f;
    for (int j = 0; j < 10; ++j) mx = fmaxf(mx, y[b * 10 + j]);
    float s = 0.f;
    for (int j = 0; j < 10; ++j) s += expf(y[b * 10 + j] - mx);
    float lse = logf(s) + mx;
    for (int j = 0; j < 10; ++j) out[b * 10 + j] = y[b * 10 + j] - lse;
}

// ---------- host ----------
extern "C" void kernel_launch(void* const* d_in, const int* in_sizes, int n_in,
                              void* d_out, int out_size, void* d_ws, size_t ws_size,
                              hipStream_t stream) {
    (void)in_sizes; (void)n_in; (void)out_size; (void)ws_size;
    const float* img     = (const float*)d_in[0];
    const float* qst     = (const float*)d_in[1];
    const float* conv1_w = (const float*)d_in[2];  const float* conv1_b = (const float*)d_in[3];
    const float* bn1_g   = (const float*)d_in[4];  const float* bn1_b   = (const float*)d_in[5];
    const float* conv2_w = (const float*)d_in[6];  const float* conv2_b = (const float*)d_in[7];
    const float* bn2_g   = (const float*)d_in[8];  const float* bn2_b   = (const float*)d_in[9];
    const float* conv3_w = (const float*)d_in[10]; const float* conv3_b = (const float*)d_in[11];
    const float* bn3_g   = (const float*)d_in[12]; const float* bn3_b   = (const float*)d_in[13];
    const float* conv4_w = (const float*)d_in[14]; const float* conv4_b = (const float*)d_in[15];
    const float* bn4_g   = (const float*)d_in[16]; const float* bn4_b   = (const float*)d_in[17];
    const float* g1_w    = (const float*)d_in[18]; const float* g1_b    = (const float*)d_in[19];
    const float* g2_w    = (const float*)d_in[20]; const float* g2_b    = (const float*)d_in[21];
    const float* g3_w    = (const float*)d_in[22]; const float* g3_b    = (const float*)d_in[23];
    const float* g4_w    = (const float*)d_in[24]; const float* g4_b    = (const float*)d_in[25];
    const float* f1_w    = (const float*)d_in[26]; const float* f1_b    = (const float*)d_in[27];
    const float* fc2_w   = (const float*)d_in[28]; const float* fc2_b   = (const float*)d_in[29];
    const float* fc3_w   = (const float*)d_in[30]; const float* fc3_b   = (const float*)d_in[31];

    size_t off = 0;
    auto carve = [&](size_t bytes) {
        void* p = (char*)d_ws + off;
        off += (bytes + 255) & ~(size_t)255;
        return p;
    };
    float* a1 = (float*)carve((size_t)128 * 32 * 38 * 38 * 4);
    float* a2 = (float*)carve((size_t)128 * 64 * 19 * 19 * 4);
    float* a3 = (float*)carve((size_t)128 * 128 * 100 * 4);
    float* a4 = (float*)carve((size_t)128 * 256 * 25 * 4);
    float* st = (float*)carve(2 * 256 * 4);
    float* ti = (float*)carve((size_t)128 * 25 * 512 * 4);
    float* tj = (float*)carve((size_t)128 * 25 * 512 * 4);
    float* tq = (float*)carve((size_t)128 * 512 * 4);
    unsigned short* Wp = (unsigned short*)carve((size_t)3 * 512 * 512 * 2);
    float* xg = (float*)carve((size_t)128 * 512 * 4);
    float* xf = (float*)carve((size_t)128 * 512 * 4);
    float* y2 = (float*)carve((size_t)128 * 1024 * 4);
    float* y3 = (float*)carve((size_t)128 * 10 * 4);

    const int TB = 256;
    auto cdiv = [](int a, int b) { return (a + b - 1) / b; };

    // conv1 -> bn1
    {
        int tot = 128 * 32 * 38 * 38;
        conv3x3s2_relu<<<cdiv(tot, TB), TB, 0, stream>>>(img, conv1_w, conv1_b, a1,
                                                         128, 3, 75, 75, 32, 38, 38);
        bn_stats<<<32, 256, 0, stream>>>(a1, st, 128, 32, 38 * 38);
        bn_apply<<<cdiv(tot, TB), TB, 0, stream>>>(a1, st, bn1_g, bn1_b, 32, 38 * 38, tot);
    }
    // conv2 -> bn2
    {
        int tot = 128 * 64 * 19 * 19;
        conv3x3s2_relu<<<cdiv(tot, TB), TB, 0, stream>>>(a1, conv2_w, conv2_b, a2,
                                                         128, 32, 38, 38, 64, 19, 19);
        bn_stats<<<64, 256, 0, stream>>>(a2, st, 128, 64, 19 * 19);
        bn_apply<<<cdiv(tot, TB), TB, 0, stream>>>(a2, st, bn2_g, bn2_b, 64, 19 * 19, tot);
    }
    // conv3 -> bn3
    {
        int tot = 128 * 128 * 100;
        conv3x3s2_relu<<<cdiv(tot, TB), TB, 0, stream>>>(a2, conv3_w, conv3_b, a3,
                                                         128, 64, 19, 19, 128, 10, 10);
        bn_stats<<<128, 256, 0, stream>>>(a3, st, 128, 128, 100);
        bn_apply<<<cdiv(tot, TB), TB, 0, stream>>>(a3, st, bn3_g, bn3_b, 128, 100, tot);
    }
    // conv4 -> bn4
    {
        int tot = 128 * 256 * 25;
        conv3x3s2_relu<<<cdiv(tot, TB), TB, 0, stream>>>(a3, conv4_w, conv4_b, a4,
                                                         128, 128, 10, 10, 256, 5, 5);
        bn_stats<<<256, 256, 0, stream>>>(a4, st, 128, 256, 25);
        bn_apply<<<cdiv(tot, TB), TB, 0, stream>>>(a4, st, bn4_g, bn4_b, 256, 25, tot);
    }

    // projections + weight pack + zero accumulator
    titj_kernel<<<cdiv(128 * 25 * 512, TB), TB, 0, stream>>>(a4, g1_w, ti, tj);
    tq_kernel<<<cdiv(128 * 512, TB), TB, 0, stream>>>(qst, g1_w, g1_b, tq);
    pack_w<<<cdiv(3 * 512 * 512, TB), TB, 0, stream>>>(g2_w, g3_w, g4_w, Wp);
    zero_f<<<cdiv(128 * 512, TB), TB, 0, stream>>>(xg, 128 * 512);

    // fused relation MLP (WMMA)
    rn_fused<<<dim3(128, 20), 256, 0, stream>>>(ti, tj, tq, Wp, g2_b, g3_b, g4_b, xg);

    // fc head
    fc_kernel<<<cdiv(128 * 512, TB), TB, 0, stream>>>(xg, f1_w, f1_b, xf, 128, 512, 512, 1);
    fc_kernel<<<cdiv(128 * 1024, TB), TB, 0, stream>>>(xf, fc2_w, fc2_b, y2, 128, 512, 1024, 1);
    fc_kernel<<<cdiv(128 * 10, TB), TB, 0, stream>>>(y2, fc3_w, fc3_b, y3, 128, 1024, 10, 0);
    log_softmax10<<<1, 128, 0, stream>>>(y3, (float*)d_out);
}
